// SelfAttention_27384711479913
// MI455X (gfx1250) — matrix operand order repaired and checked by reading, not compiled, not run on hardware
//
#include <hip/hip_runtime.h>
#include <math.h>

// Problem constants (from reference): B=32, L=512, E=256, H=8, D=64
#define B_ 32
#define L_ 512
#define E_ 256
#define H_ 8
#define D_ 64
#define F_ 512            // H*D
#define SCL 0.125f        // 1 / int(sqrt(D)) = 1/8
#define LOG2E_ 1.4426950408889634f

typedef __attribute__((ext_vector_type(16))) __bf16 bf16x16;
typedef __attribute__((ext_vector_type(8)))  __bf16 bf16x8;
typedef __attribute__((ext_vector_type(8)))  float  f32x8;

#define AS1 __attribute__((address_space(1)))
#define AS3 __attribute__((address_space(3)))

// Pointee type required by the async-copy builtins (per hipcc diagnostic):
// '__attribute__((__vector_size__(4 * sizeof(int)))) int __device__ *'
typedef __attribute__((__vector_size__(4 * sizeof(int)))) int v4i_;

// CDNA5 async global<->LDS path (ASYNCcnt-tracked), with sync fallback.
#if __has_builtin(__builtin_amdgcn_global_load_async_to_lds_b128) && \
    __has_builtin(__builtin_amdgcn_s_wait_asynccnt)
#define USE_ASYNC_LDS 1
#else
#define USE_ASYNC_LDS 0
#endif

#if __has_builtin(__builtin_amdgcn_global_store_async_from_lds_b128)
#define USE_ASYNC_STORE 1
#else
#define USE_ASYNC_STORE 0
#endif

#if USE_ASYNC_LDS
__device__ __forceinline__ void async_g2l_16B(void* lds, const void* g) {
  __builtin_amdgcn_global_load_async_to_lds_b128((AS1 v4i_*)g, (AS3 v4i_*)lds, 0, 0);
}
__device__ __forceinline__ void wait_async0() {
  __builtin_amdgcn_s_wait_asynccnt(0);
}
#endif

// ---------------------------------------------------------------- intrinsics
__device__ __forceinline__ float act_tanh(float x) {
#if __has_builtin(__builtin_amdgcn_tanhf)
  return __builtin_amdgcn_tanhf(x);   // gfx1250 V_TANH_F32
#else
  return tanhf(x);
#endif
}

__device__ __forceinline__ float exp2_fast(float x) {
#if __has_builtin(__builtin_amdgcn_exp2f)
  return __builtin_amdgcn_exp2f(x);
#else
  return exp2f(x);
#endif
}

__device__ __forceinline__ f32x8 wmma_bf16(bf16x16 a, bf16x16 b, f32x8 c) {
  // (neg_a, A, neg_b, B, c_mod, C, reuse_a, reuse_b)
  return __builtin_amdgcn_wmma_f32_16x16x32_bf16(false, a, false, b, (short)0, c,
                                                 false, false);
}

// A fragment (16x32, M x K), src row-major with leading dim `ld` (K contiguous).
// ISA layout: lanes 0-15: row M=lane, K {0..7}+{16..23}; lanes 16-31: K {8..15}+{24..31}.
__device__ __forceinline__ bf16x16 load_frag_a(const __bf16* base, int ld, int m0,
                                               int k0, int lane) {
  const int r = lane & 15, hi = lane >> 4;
  const __bf16* p = base + (size_t)(m0 + r) * ld + k0 + hi * 8;
  bf16x8 lo = *(const bf16x8*)(p);
  bf16x8 hh = *(const bf16x8*)(p + 16);
  bf16x16 f;
#pragma unroll
  for (int i = 0; i < 8; ++i) { f[i] = lo[i]; f[i + 8] = hh[i]; }
  return f;
}

// B fragment (32x16, K x N), given B^T row-major [N x K] with leading dim `ld`.
// Layout: lanes 0-15: col N=lane, K {0..7}+{16..23}; lanes 16-31: K {8..15}+{24..31}.
__device__ __forceinline__ bf16x16 load_frag_b(const __bf16* baseT, int ld, int n0,
                                               int k0, int lane) {
  const int r = lane & 15, hi = lane >> 4;
  const __bf16* p = baseT + (size_t)(n0 + r) * ld + k0 + hi * 8;
  bf16x8 lo = *(const bf16x8*)(p);
  bf16x8 hh = *(const bf16x8*)(p + 16);
  bf16x16 f;
#pragma unroll
  for (int i = 0; i < 8; ++i) { f[i] = lo[i]; f[i + 8] = hh[i]; }
  return f;
}

// ---------------------------------------------------------------- kernel 1: prep
// xp = x + pos (bf16); residual tanh(xp) (f32) -> d_out
__global__ __launch_bounds__(256) void prep_kernel(const float* __restrict__ x,
                                                   const float* __restrict__ pos,
                                                   __bf16* __restrict__ xp,
                                                   float* __restrict__ res) {
  int i = blockIdx.x * 256 + threadIdx.x;           // exact: B*L*E / 256 blocks
  int le = i % (L_ * E_);
  float v = x[i] + pos[le];
  xp[i] = (__bf16)v;
  res[i] = act_tanh(v);
}

// ------------------------------------------------- kernel 2: weight transposes
// [H,E,D] f32 -> [H,D,E] bf16  (makes B-fragment loads K-contiguous)
__global__ __launch_bounds__(256) void wT_kernel(const float* __restrict__ W,
                                                 __bf16* __restrict__ WT) {
  int i = blockIdx.x * 256 + threadIdx.x;           // H*E*D threads
  int h = i / (E_ * D_), rr = i % (E_ * D_);
  int d = rr / E_, e = rr % E_;
  WT[i] = (__bf16)W[(size_t)h * E_ * D_ + (size_t)e * D_ + d];
}

// Wo [F,E] f32 -> WoT [E,F] bf16
__global__ __launch_bounds__(256) void woT_kernel(const float* __restrict__ Wo,
                                                  __bf16* __restrict__ WoT) {
  int i = blockIdx.x * 256 + threadIdx.x;           // E*F threads
  int e = i / F_, f = i % F_;
  WoT[i] = (__bf16)Wo[(size_t)f * E_ + e];
}

// ------------------------------------------------ kernel 3: q/k/v projections
// Per (b,h): tanh(xp[L,E] @ W[E,D]).  q,k stored [L,D]; v stored transposed [D,L].
__global__ __launch_bounds__(256) void proj_kernel(
    const __bf16* __restrict__ xp, const __bf16* __restrict__ WqT,
    const __bf16* __restrict__ WkT, const __bf16* __restrict__ WvT,
    __bf16* __restrict__ qg, __bf16* __restrict__ kg, __bf16* __restrict__ vTg) {
  __shared__ __bf16 sbuf[8][16][72];                // per-wave transpose staging
  const int b = blockIdx.z, h = blockIdx.y;
  const int w = threadIdx.x >> 5, lane = threadIdx.x & 31;
  const int m0 = blockIdx.x * 128 + w * 16;
  const int r = lane & 15, hi = lane >> 4;

  const __bf16* A   = xp  + (size_t)b * L_ * E_;
  const __bf16* Wt0 = WqT + (size_t)h * D_ * E_;
  const __bf16* Wt1 = WkT + (size_t)h * D_ * E_;
  const __bf16* Wt2 = WvT + (size_t)h * D_ * E_;

  f32x8 acc[3][4];
#pragma unroll
  for (int m = 0; m < 3; ++m)
#pragma unroll
    for (int t = 0; t < 4; ++t) acc[m][t] = (f32x8){0, 0, 0, 0, 0, 0, 0, 0};

  for (int k0 = 0; k0 < E_; k0 += 32) {
    bf16x16 a = load_frag_a(A, E_, m0, k0, lane);
#pragma unroll
    for (int t = 0; t < 4; ++t) {
      acc[0][t] = wmma_bf16(a, load_frag_b(Wt0, E_, t * 16, k0, lane), acc[0][t]);
      acc[1][t] = wmma_bf16(a, load_frag_b(Wt1, E_, t * 16, k0, lane), acc[1][t]);
      acc[2][t] = wmma_bf16(a, load_frag_b(Wt2, E_, t * 16, k0, lane), acc[2][t]);
    }
  }

  const size_t bh = (size_t)b * H_ + h;

  // V: C-layout lane holds (fixed d, 8 consecutive l) -> direct coalesced store to vT[D,L]
#pragma unroll
  for (int t = 0; t < 4; ++t) {
    bf16x8 pk;
#pragma unroll
    for (int j = 0; j < 8; ++j) pk[j] = (__bf16)act_tanh(acc[2][t][j]);
    __bf16* dst = vTg + bh * (size_t)(D_ * L_) + (size_t)(t * 16 + r) * L_ + m0 + hi * 8;
    *(bf16x8*)dst = pk;
  }

  // Q, K: transpose through LDS -> row-major [L,D] with coalesced b128 stores
  __bf16* outs[2] = { qg + bh * (size_t)(L_ * D_), kg + bh * (size_t)(L_ * D_) };
  for (int m = 0; m < 2; ++m) {
    __syncthreads();
#pragma unroll
    for (int t = 0; t < 4; ++t)
#pragma unroll
      for (int j = 0; j < 8; ++j)
        sbuf[w][j + 8 * hi][t * 16 + r] = (__bf16)act_tanh(acc[m][t][j]);
    __syncthreads();
    __bf16* dst = outs[m] + (size_t)(m0 + r) * D_ + hi * 32;
    const __bf16* src = &sbuf[w][r][hi * 32];
#pragma unroll
    for (int i = 0; i < 4; ++i)
      *(bf16x8*)(dst + i * 8) = *(const bf16x8*)(src + i * 8);
  }
}

// ---------------------------------------------- kernel 4: flash attention core
// Per (b,h): O = softmax(q k^T / 8) v ; O stored [B, L, F] (head-concatenated)
__global__ __launch_bounds__(256) void attn_kernel(const __bf16* __restrict__ qg,
                                                   const __bf16* __restrict__ kg,
                                                   const __bf16* __restrict__ vTg,
                                                   __bf16* __restrict__ Og) {
  __shared__ __bf16 sK[64][72];      // K block  [kk][d]
  __shared__ __bf16 sV[64][72];      // V block  [d][kk]   (from vT)
  __shared__ __bf16 sP[8][16][72];   // per-wave P / O staging [lq][*]
  const int b = blockIdx.z, h = blockIdx.y;
  const int w = threadIdx.x >> 5, lane = threadIdx.x & 31;
  const int m0 = blockIdx.x * 128 + w * 16;
  const int r = lane & 15, hi = lane >> 4;
  const size_t bh = (size_t)b * H_ + h;

  const __bf16* qb = qg + bh * (size_t)(L_ * D_);
  const __bf16* kb = kg + bh * (size_t)(L_ * D_);
  const __bf16* vb = vTg + bh * (size_t)(D_ * L_);

  bf16x16 qf0 = load_frag_a(qb, D_, m0, 0, lane);
  bf16x16 qf1 = load_frag_a(qb, D_, m0, 32, lane);

  f32x8 Oacc[4];
#pragma unroll
  for (int t = 0; t < 4; ++t) Oacc[t] = (f32x8){0, 0, 0, 0, 0, 0, 0, 0};
  float mrow[8], lrow[8];
#pragma unroll
  for (int j = 0; j < 8; ++j) { mrow[j] = -1e30f; lrow[j] = 0.f; }

  for (int kb0 = 0; kb0 < L_; kb0 += 64) {
    __syncthreads();
    // Cooperative staging of 64x64 bf16 K-tile and V-tile (512 x 16B chunks each).
#if USE_ASYNC_LDS
    // CDNA5 async copy: global -> LDS without touching VGPRs (ASYNCcnt-tracked).
    for (int c = threadIdx.x; c < 512; c += 256) {
      int row = c >> 3, cc = c & 7;
      async_g2l_16B(&sK[row][cc * 8], kb + (size_t)(kb0 + row) * D_ + cc * 8);
      async_g2l_16B(&sV[row][cc * 8], vb + (size_t)row * L_ + kb0 + cc * 8);
    }
    wait_async0();
#else
    for (int c = threadIdx.x; c < 512; c += 256) {
      int row = c >> 3, cc = c & 7;
      *(bf16x8*)(&sK[row][cc * 8]) =
          *(const bf16x8*)(kb + (size_t)(kb0 + row) * D_ + cc * 8);
      *(bf16x8*)(&sV[row][cc * 8]) =
          *(const bf16x8*)(vb + (size_t)row * L_ + kb0 + cc * 8);
    }
#endif
    if (kb0 + 64 < L_) {  // prefetch next block (global_prefetch_b8)
      int row = threadIdx.x >> 2, seg = threadIdx.x & 3;
      __builtin_prefetch(kb + (size_t)(kb0 + 64 + (threadIdx.x >> 1)) * D_ +
                             (threadIdx.x & 1) * 32, 0, 0);
      __builtin_prefetch(vb + (size_t)row * L_ + kb0 + 64 + seg * 16, 0, 0);
    }
    __syncthreads();

    // S = q k^T  (4 tiles of 16x16, K over d = 2 steps)
    f32x8 S[4];
#pragma unroll
    for (int t = 0; t < 4; ++t) {
      S[t] = (f32x8){0, 0, 0, 0, 0, 0, 0, 0};
      S[t] = wmma_bf16(qf0, load_frag_b(&sK[0][0], 72, t * 16, 0, lane), S[t]);
      S[t] = wmma_bf16(qf1, load_frag_b(&sK[0][0], 72, t * 16, 32, lane), S[t]);
    }

    // Online softmax (row j lives in one VGPR across a 16-lane half)
#pragma unroll
    for (int j = 0; j < 8; ++j) {
      float tm = -1e30f;
#pragma unroll
      for (int t = 0; t < 4; ++t) { S[t][j] *= SCL; tm = fmaxf(tm, S[t][j]); }
#pragma unroll
      for (int mk = 1; mk < 16; mk <<= 1) tm = fmaxf(tm, __shfl_xor(tm, mk, 32));
      float nm = fmaxf(mrow[j], tm);
      float alpha = exp2_fast((mrow[j] - nm) * LOG2E_);
      mrow[j] = nm;
      float rs = 0.f;
#pragma unroll
      for (int t = 0; t < 4; ++t) {
        float p = exp2_fast((S[t][j] - nm) * LOG2E_);
        S[t][j] = p;
        rs += p;
      }
#pragma unroll
      for (int mk = 1; mk < 16; mk <<= 1) rs += __shfl_xor(rs, mk, 32);
      lrow[j] = lrow[j] * alpha + rs;
#pragma unroll
      for (int t = 0; t < 4; ++t) Oacc[t][j] *= alpha;
    }

    // P -> wave-private LDS, re-read as A fragments
#pragma unroll
    for (int t = 0; t < 4; ++t)
#pragma unroll
      for (int j = 0; j < 8; ++j)
        sP[w][j + 8 * hi][t * 16 + r] = (__bf16)S[t][j];

    bf16x16 pf0 = load_frag_a(&sP[w][0][0], 72, 0, 0, lane);
    bf16x16 pf1 = load_frag_a(&sP[w][0][0], 72, 0, 32, lane);
#pragma unroll
    for (int t = 0; t < 4; ++t) {
      Oacc[t] = wmma_bf16(pf0, load_frag_b(&sV[0][0], 72, t * 16, 0, lane), Oacc[t]);
      Oacc[t] = wmma_bf16(pf1, load_frag_b(&sV[0][0], 72, t * 16, 32, lane), Oacc[t]);
    }
  }

  // Normalize and write O[B,L,F] (transpose via LDS, coalesced b128 stores)
#pragma unroll
  for (int j = 0; j < 8; ++j) {
    float inv = 1.0f / lrow[j];
#pragma unroll
    for (int t = 0; t < 4; ++t) Oacc[t][j] *= inv;
  }
  __syncthreads();
#pragma unroll
  for (int t = 0; t < 4; ++t)
#pragma unroll
    for (int j = 0; j < 8; ++j)
      sP[w][j + 8 * hi][t * 16 + r] = (__bf16)Oacc[t][j];
  __syncthreads();
  __bf16* dst = Og + ((size_t)b * L_ + m0 + r) * F_ + h * D_ + hi * 32;
  const __bf16* src = &sP[w][r][hi * 32];
#if USE_ASYNC_LDS && USE_ASYNC_STORE
  // LDS -> global without a VGPR round-trip; explicit asynccnt wait (plus
  // S_ENDPGM's implicit wait-idle) guarantees completion.
#pragma unroll
  for (int i = 0; i < 4; ++i)
    __builtin_amdgcn_global_store_async_from_lds_b128(
        (AS1 v4i_*)(dst + i * 8), (AS3 v4i_*)(src + i * 8), 0, 0);
  wait_async0();
#else
#pragma unroll
  for (int i = 0; i < 4; ++i)
    *(bf16x8*)(dst + i * 8) = *(const bf16x8*)(src + i * 8);
#endif
}

// --------------------------------------------- kernel 5: output projection
// out[B*L, E] = O[B*L, F] @ Wo[F, E] + residual (already in d_out)
__global__ __launch_bounds__(256) void outproj_kernel(const __bf16* __restrict__ O,
                                                      const __bf16* __restrict__ WoT,
                                                      float* __restrict__ out) {
  const int w = threadIdx.x >> 5, lane = threadIdx.x & 31;
  const int m0 = blockIdx.x * 128 + w * 16;   // rows over B*L
  const int n0b = blockIdx.y * 64;            // 64-wide column block of E
  const int r = lane & 15, hi = lane >> 4;

  f32x8 acc[4];
#pragma unroll
  for (int t = 0; t < 4; ++t) acc[t] = (f32x8){0, 0, 0, 0, 0, 0, 0, 0};

  for (int k0 = 0; k0 < F_; k0 += 32) {
    bf16x16 a = load_frag_a(O, F_, m0, k0, lane);
#pragma unroll
    for (int t = 0; t < 4; ++t)
      acc[t] = wmma_bf16(a, load_frag_b(WoT, F_, n0b + t * 16, k0, lane), acc[t]);
  }

#pragma unroll
  for (int t = 0; t < 4; ++t) {
    int e = n0b + t * 16 + r;
#pragma unroll
    for (int j = 0; j < 8; ++j) {
      size_t row = (size_t)m0 + j + 8 * hi;
      out[row * E_ + e] += acc[t][j];       // residual tanh(xp) pre-stored by prep
    }
  }
}

// ------------------------------------------------------------------ launcher
extern "C" void kernel_launch(void* const* d_in, const int* in_sizes, int n_in,
                              void* d_out, int out_size, void* d_ws, size_t ws_size,
                              hipStream_t stream) {
  (void)in_sizes; (void)n_in; (void)out_size; (void)ws_size;
  const float* x   = (const float*)d_in[0];
  const float* pos = (const float*)d_in[1];
  const float* Wq  = (const float*)d_in[2];
  const float* Wk  = (const float*)d_in[3];
  const float* Wv  = (const float*)d_in[4];
  const float* Wo  = (const float*)d_in[5];
  float* out = (float*)d_out;

  char* ws = (char*)d_ws;
  size_t off = 0;
  auto take = [&](size_t bytes) -> char* {
    char* p = ws + off;
    off += (bytes + 255) & ~(size_t)255;
    return p;
  };
  __bf16* xp  = (__bf16*)take((size_t)B_ * L_ * E_ * 2);
  __bf16* WqT = (__bf16*)take((size_t)H_ * D_ * E_ * 2);
  __bf16* WkT = (__bf16*)take((size_t)H_ * D_ * E_ * 2);
  __bf16* WvT = (__bf16*)take((size_t)H_ * D_ * E_ * 2);
  __bf16* WoT = (__bf16*)take((size_t)E_ * F_ * 2);
  __bf16* qb  = (__bf16*)take((size_t)B_ * H_ * L_ * D_ * 2);
  __bf16* kb  = (__bf16*)take((size_t)B_ * H_ * L_ * D_ * 2);
  __bf16* vT  = (__bf16*)take((size_t)B_ * H_ * D_ * L_ * 2);
  __bf16* Ob  = (__bf16*)take((size_t)B_ * L_ * F_ * 2);

  prep_kernel<<<(B_ * L_ * E_) / 256, 256, 0, stream>>>(x, pos, xp, out);
  wT_kernel<<<(H_ * E_ * D_) / 256, 256, 0, stream>>>(Wq, WqT);
  wT_kernel<<<(H_ * E_ * D_) / 256, 256, 0, stream>>>(Wk, WkT);
  wT_kernel<<<(H_ * E_ * D_) / 256, 256, 0, stream>>>(Wv, WvT);
  woT_kernel<<<(E_ * F_) / 256, 256, 0, stream>>>(Wo, WoT);
  proj_kernel<<<dim3(L_ / 128, H_, B_), 256, 0, stream>>>(xp, WqT, WkT, WvT, qb, kb, vT);
  attn_kernel<<<dim3(L_ / 128, H_, B_), 256, 0, stream>>>(qb, kb, vT, Ob);
  outproj_kernel<<<dim3((B_ * L_) / 128, E_ / 64), 256, 0, stream>>>(Ob, WoT, out);
}
